// Mamba2_72000831750785
// MI455X (gfx1250) — matrix operand order repaired and checked by reading, not compiled, not run on hardware
//
#include <hip/hip_runtime.h>
#include <hip/hip_bf16.h>
#include <math.h>

// ---------------------------------------------------------------------------
// Mamba2 forward for MI455X (gfx1250), wave32 + WMMA bf16.
// GEMMs consume pre-swizzled bf16 operands in exact WMMA fragment order;
// hot loop = global_load_b128 + v_wmma with 4x4 register blocking.
// ---------------------------------------------------------------------------

typedef __attribute__((ext_vector_type(16))) __bf16 v16bf;
typedef __attribute__((ext_vector_type(8)))  float  v8f;

#define L_SEQ    2048
#define D_MODEL  1024
#define D_INNER  2048
#define D_XB     512
#define D_STATE  16
#define DT_RANK  64
#define D_PROJ   (2*D_INNER + 2*D_XB + DT_RANK)   // 5184
#define G_HEADS  (D_INNER / D_STATE)              // 128
#define NCHUNK   16
#define LCHUNK   (L_SEQ / NCHUNK)                 // 128

// column offsets inside zxbcdt
#define OFF_Z   0
#define OFF_X   D_INNER                  // 2048
#define OFF_B   (D_INNER + D_XB)         // 2560
#define OFF_C   (D_INNER + 2*D_XB)       // 3072
#define OFF_DT  (2*D_INNER + 2*D_XB)     // 5120

// fragment-swizzled buffer sizes (bf16 elements): tiles * kTiles * 32 lanes * 16
#define SZ_ASW1  ((size_t)(L_SEQ/16)   * (D_MODEL/32) * 512)
#define SZ_BSW1  ((size_t)(D_PROJ/16)  * (D_MODEL/32) * 512)
#define SZ_ASWDT ((size_t)(L_SEQ/16)   * (DT_RANK/32) * 512)
#define SZ_BSWDT ((size_t)(D_INNER/16) * (DT_RANK/32) * 512)
#define SZ_ASW2  ((size_t)(L_SEQ/16)   * (D_INNER/32) * 512)
#define SZ_BSW2  ((size_t)(D_MODEL/16) * (D_INNER/32) * 512)

// ---------------------------------------------------------------------------
// Swizzle f32 A (M x K row-major) -> bf16 fragment order
// ---------------------------------------------------------------------------
__global__ void __launch_bounds__(256)
swizzleA_bf16(const float* __restrict__ A, __bf16* __restrict__ out,
              int mTiles, int kTiles, int lda)
{
    int idx = blockIdx.x * blockDim.x + threadIdx.x;
    if (idx >= mTiles * kTiles * 512) return;
    int e    = idx & 15;
    int lane = (idx >> 4) & 31;
    int kt   = (idx >> 9) % kTiles;
    int mt   = (idx >> 9) / kTiles;
    int kk   = ((e >> 3) << 4) + ((lane >> 4) << 3) + (e & 7);
    int m    = (mt << 4) + (lane & 15);
    int k    = (kt << 5) + kk;
    out[idx] = (__bf16)A[(long)m * lda + k];
}

// ---------------------------------------------------------------------------
// Swizzle f32 B (K x N row-major) -> bf16 fragment order
// ---------------------------------------------------------------------------
__global__ void __launch_bounds__(256)
swizzleB_bf16(const float* __restrict__ B, __bf16* __restrict__ out,
              int nTiles, int kTiles, int ldb)
{
    int idx = blockIdx.x * blockDim.x + threadIdx.x;
    if (idx >= nTiles * kTiles * 512) return;
    int e    = idx & 15;
    int lane = (idx >> 4) & 31;
    int kt   = (idx >> 9) % kTiles;
    int nt   = (idx >> 9) / kTiles;
    int kk   = ((e >> 3) << 4) + ((lane >> 4) << 3) + (e & 7);
    int n    = (nt << 4) + (lane & 15);
    int k    = (kt << 5) + kk;
    out[idx] = (__bf16)B[(long)k * ldb + n];
}

// ---------------------------------------------------------------------------
// Register-blocked WMMA GEMM on swizzled bf16 operands.
// Block = 128 threads (4 waves stacked on M) -> tile 256(M) x 64(N).
// Wave tile = 64x64: 4 A frags x 4 B frags -> 16 WMMAs per K-step against
// 16 b128 loads (32 FLOP/byte from L2). All pointers are scalar locals so
// loads lower to global_load_b128 (not flat).
// Requires M%256==0, N%64==0, K%32==0.
// ---------------------------------------------------------------------------
__global__ void __launch_bounds__(128)
gemm_bf16_wmma(const __bf16* __restrict__ Asw, const __bf16* __restrict__ Bsw,
               float* __restrict__ C, int kTiles, int ldc)
{
    const int lane = threadIdx.x & 31;
    const int wave = threadIdx.x >> 5;                 // 0..3 -> M sub-range
    const int mtBase = blockIdx.x * 16 + wave * 4;     // 16x16-row tiles / block
    const long fragStride = (long)kTiles * 512;        // elems between tiles

    const __bf16* pA0 = Asw + (mtBase + 0) * fragStride + lane * 16;
    const __bf16* pA1 = pA0 + fragStride;
    const __bf16* pA2 = pA1 + fragStride;
    const __bf16* pA3 = pA2 + fragStride;

    const long nBaseT = (long)blockIdx.y * 4;
    const __bf16* pB0 = Bsw + (nBaseT + 0) * fragStride + lane * 16;
    const __bf16* pB1 = pB0 + fragStride;
    const __bf16* pB2 = pB1 + fragStride;
    const __bf16* pB3 = pB2 + fragStride;

    v8f acc[4][4] = {};

    for (int kt = 0; kt < kTiles; ++kt) {
        v16bf a0 = *(const v16bf*)pA0;  __builtin_prefetch(pA0 + 512, 0, 1);
        v16bf a1 = *(const v16bf*)pA1;  __builtin_prefetch(pA1 + 512, 0, 1);
        v16bf a2 = *(const v16bf*)pA2;  __builtin_prefetch(pA2 + 512, 0, 1);
        v16bf a3 = *(const v16bf*)pA3;  __builtin_prefetch(pA3 + 512, 0, 1);
        v16bf b0 = *(const v16bf*)pB0;  __builtin_prefetch(pB0 + 512, 0, 1);
        v16bf b1 = *(const v16bf*)pB1;  __builtin_prefetch(pB1 + 512, 0, 1);
        v16bf b2 = *(const v16bf*)pB2;  __builtin_prefetch(pB2 + 512, 0, 1);
        v16bf b3 = *(const v16bf*)pB3;  __builtin_prefetch(pB3 + 512, 0, 1);
        pA0 += 512; pA1 += 512; pA2 += 512; pA3 += 512;
        pB0 += 512; pB1 += 512; pB2 += 512; pB3 += 512;

#define WMMA_ROW(i, ai)                                                        \
        acc[i][0] = __builtin_amdgcn_wmma_f32_16x16x32_bf16(                   \
                        false, ai, false, b0, (short)0, acc[i][0], false, false); \
        acc[i][1] = __builtin_amdgcn_wmma_f32_16x16x32_bf16(                   \
                        false, ai, false, b1, (short)0, acc[i][1], false, false); \
        acc[i][2] = __builtin_amdgcn_wmma_f32_16x16x32_bf16(                   \
                        false, ai, false, b2, (short)0, acc[i][2], false, false); \
        acc[i][3] = __builtin_amdgcn_wmma_f32_16x16x32_bf16(                   \
                        false, ai, false, b3, (short)0, acc[i][3], false, false);
        WMMA_ROW(0, a0)
        WMMA_ROW(1, a1)
        WMMA_ROW(2, a2)
        WMMA_ROW(3, a3)
#undef WMMA_ROW
    }

    // D layout: lane n = lane&15, rows m = vgpr + (lane<16 ? 0 : 8)
    const int rOff = (lane < 16) ? 0 : 8;
#pragma unroll
    for (int i = 0; i < 4; ++i) {
        const int mBase = (mtBase + i) * 16 + rOff;
#pragma unroll
        for (int j = 0; j < 4; ++j) {
            const int n = blockIdx.y * 64 + j * 16 + (lane & 15);
#pragma unroll
            for (int r = 0; r < 8; ++r)
                C[(long)(mBase + r) * ldc + n] = acc[i][j][r];
        }
    }
}

// ---------------------------------------------------------------------------
// Causal depthwise conv (K=4) + bias + SiLU over the x slice of zxbcdt.
// ---------------------------------------------------------------------------
__global__ void __launch_bounds__(256)
conv_silu_kernel(const float* __restrict__ zxbcdt,
                 const float* __restrict__ conv_w,
                 const float* __restrict__ conv_b,
                 float* __restrict__ xout)
{
    int idx = blockIdx.x * blockDim.x + threadIdx.x;   // l * D_XB + d
    if (idx >= L_SEQ * D_XB) return;
    int l = idx >> 9;
    int d = idx & (D_XB - 1);
    float acc = conv_b[d];
#pragma unroll
    for (int k = 0; k < 4; ++k) {
        int ll = l - 3 + k;
        if (ll >= 0)
            acc += zxbcdt[(long)ll * D_PROJ + OFF_X + d] * conv_w[d * 4 + k];
    }
    xout[idx] = acc / (1.0f + __expf(-acc));           // silu
}

// ---------------------------------------------------------------------------
// delta = softplus(delta_raw + 2*dt_bias)   (in place, faithful double-bias)
// ---------------------------------------------------------------------------
__global__ void __launch_bounds__(256)
dt_softplus_kernel(float* __restrict__ delta, const float* __restrict__ dt_bias)
{
    int idx = blockIdx.x * blockDim.x + threadIdx.x;   // l * D_INNER + d
    if (idx >= L_SEQ * D_INNER) return;
    int d = idx & (D_INNER - 1);
    float v = delta[idx] + 2.0f * dt_bias[d];
    delta[idx] = (v > 20.0f) ? v : log1pf(__expf(v));
}

// ---------------------------------------------------------------------------
// Chunked selective scan, pass 1: per-chunk carries (prod a, local h).
// ---------------------------------------------------------------------------
__global__ void __launch_bounds__(256)
scan_pass1(const float* __restrict__ zxbcdt, const float* __restrict__ xconv,
           const float* __restrict__ delta,  const float* __restrict__ A_log,
           float* __restrict__ carry_a, float* __restrict__ carry_b)
{
    int g = blockIdx.x, c = blockIdx.y;
    int p = threadIdx.x >> 4, n = threadIdx.x & 15;
    int dcol = g * D_STATE + p;
    int xcol = (g >> 2) * D_STATE + p;
    int bcol = OFF_B + (g >> 2) * D_STATE + n;
    float Apn = -__expf(A_log[dcol * D_STATE + n]);
    float aprod = 1.0f, h = 0.0f;
    int l0 = c * LCHUNK;
    for (int l = l0; l < l0 + LCHUNK; ++l) {
        float dlt = delta[l * D_INNER + dcol];
        float xv  = xconv[l * D_XB + xcol];
        float Bv  = zxbcdt[(long)l * D_PROJ + bcol];
        float a   = __expf(dlt * Apn);
        h = fmaf(a, h, dlt * xv * Bv);
        aprod *= a;
    }
    int o = ((g * NCHUNK + c) * D_STATE + p) * D_STATE + n;
    carry_a[o] = aprod;
    carry_b[o] = h;
}

// ---------------------------------------------------------------------------
// Pass 2: sequential combine of the 16 chunk carries -> per-chunk init state.
// ---------------------------------------------------------------------------
__global__ void __launch_bounds__(256)
scan_pass2(const float* __restrict__ carry_a, const float* __restrict__ carry_b,
           float* __restrict__ init)
{
    int g = blockIdx.x;
    int p = threadIdx.x >> 4, n = threadIdx.x & 15;
    float s = 0.0f;
    for (int c = 0; c < NCHUNK; ++c) {
        int o = ((g * NCHUNK + c) * D_STATE + p) * D_STATE + n;
        init[o] = s;
        s = carry_a[o] * s + carry_b[o];
    }
}

// ---------------------------------------------------------------------------
// Pass 3: re-scan with init state; fuse y = sum_n h*C (wave32 shfl_xor over
// the 16 n-lanes), + D*x skip, * silu(z) gate.
// ---------------------------------------------------------------------------
__global__ void __launch_bounds__(256)
scan_pass3(const float* __restrict__ zxbcdt, const float* __restrict__ xconv,
           const float* __restrict__ delta,  const float* __restrict__ A_log,
           const float* __restrict__ Dvec,   const float* __restrict__ init,
           float* __restrict__ ygated)
{
    int g = blockIdx.x, c = blockIdx.y;
    int p = threadIdx.x >> 4, n = threadIdx.x & 15;
    int dcol = g * D_STATE + p;
    int xcol = (g >> 2) * D_STATE + p;
    int bcol = OFF_B + (g >> 2) * D_STATE + n;
    int ccol = OFF_C + g * D_STATE + n;
    float Apn = -__expf(A_log[dcol * D_STATE + n]);
    float h   = init[((g * NCHUNK + c) * D_STATE + p) * D_STATE + n];
    float Dp  = Dvec[dcol];
    int l0 = c * LCHUNK;
    for (int l = l0; l < l0 + LCHUNK; ++l) {
        long base = (long)l * D_PROJ;
        float dlt = delta[l * D_INNER + dcol];
        float xv  = xconv[l * D_XB + xcol];
        float Bv  = zxbcdt[base + bcol];
        float Cv  = zxbcdt[base + ccol];
        float a   = __expf(dlt * Apn);
        h = fmaf(a, h, dlt * xv * Bv);
        float y = h * Cv;
        y += __shfl_xor(y, 1, 16);
        y += __shfl_xor(y, 2, 16);
        y += __shfl_xor(y, 4, 16);
        y += __shfl_xor(y, 8, 16);
        if (n == 0) {
            float z  = zxbcdt[base + OFF_Z + dcol];
            float sz = z / (1.0f + __expf(-z));
            ygated[l * D_INNER + dcol] = (y + Dp * xv) * sz;
        }
    }
}

// ---------------------------------------------------------------------------
// Launch
// ---------------------------------------------------------------------------
extern "C" void kernel_launch(void* const* d_in, const int* in_sizes, int n_in,
                              void* d_out, int out_size, void* d_ws, size_t ws_size,
                              hipStream_t stream)
{
    const float* hidden  = (const float*)d_in[0];
    const float* W_in    = (const float*)d_in[1];
    const float* conv_w  = (const float*)d_in[2];
    const float* conv_b  = (const float*)d_in[3];
    const float* W_dt    = (const float*)d_in[4];
    const float* dt_bias = (const float*)d_in[5];
    const float* A_log   = (const float*)d_in[6];
    const float* Dvec    = (const float*)d_in[7];
    const float* W_out   = (const float*)d_in[8];
    float* out = (float*)d_out;

    // ---- workspace carve-up: f32 region then bf16 (swizzled) region ----
    float* ws      = (float*)d_ws;
    float* zxbcdt  = ws;
    float* xconv   = zxbcdt + (size_t)L_SEQ * D_PROJ;
    float* delta   = xconv  + (size_t)L_SEQ * D_XB;
    float* carry_a = delta  + (size_t)L_SEQ * D_INNER;
    float* carry_b = carry_a + (size_t)G_HEADS * NCHUNK * D_STATE * D_STATE;
    float* initst  = carry_b + (size_t)G_HEADS * NCHUNK * D_STATE * D_STATE;
    float* ygated  = initst  + (size_t)G_HEADS * NCHUNK * D_STATE * D_STATE;

    __bf16* Asw1  = (__bf16*)(ygated + (size_t)L_SEQ * D_INNER);
    __bf16* Bsw1  = Asw1  + SZ_ASW1;
    __bf16* AswDT = Bsw1  + SZ_BSW1;
    __bf16* BswDT = AswDT + SZ_ASWDT;
    __bf16* Asw2  = BswDT + SZ_BSWDT;
    __bf16* Bsw2  = Asw2  + SZ_ASW2;

    // 1) zxbcdt = hidden @ W_in   (2048 x 1024 x 5184)
    swizzleA_bf16<<<(int)(SZ_ASW1 + 255) / 256, 256, 0, stream>>>(
        hidden, Asw1, L_SEQ/16, D_MODEL/32, D_MODEL);
    swizzleB_bf16<<<(int)(SZ_BSW1 + 255) / 256, 256, 0, stream>>>(
        W_in, Bsw1, D_PROJ/16, D_MODEL/32, D_PROJ);
    gemm_bf16_wmma<<<dim3(L_SEQ/256, D_PROJ/64), 128, 0, stream>>>(
        Asw1, Bsw1, zxbcdt, D_MODEL/32, D_PROJ);

    // 2) causal depthwise conv + silu on x slice
    conv_silu_kernel<<<(L_SEQ*D_XB + 255)/256, 256, 0, stream>>>(
        zxbcdt, conv_w, conv_b, xconv);

    // 3) delta_raw = dtr @ W_dt   (2048 x 64 x 2048)
    swizzleA_bf16<<<(int)(SZ_ASWDT + 255) / 256, 256, 0, stream>>>(
        zxbcdt + OFF_DT, AswDT, L_SEQ/16, DT_RANK/32, D_PROJ);
    swizzleB_bf16<<<(int)(SZ_BSWDT + 255) / 256, 256, 0, stream>>>(
        W_dt, BswDT, D_INNER/16, DT_RANK/32, D_INNER);
    gemm_bf16_wmma<<<dim3(L_SEQ/256, D_INNER/64), 128, 0, stream>>>(
        AswDT, BswDT, delta, DT_RANK/32, D_INNER);

    // 4) delta = softplus(delta_raw + 2*dt_bias)
    dt_softplus_kernel<<<(L_SEQ*D_INNER + 255)/256, 256, 0, stream>>>(delta, dt_bias);

    // 5-7) chunked selective scan
    scan_pass1<<<dim3(G_HEADS, NCHUNK), 256, 0, stream>>>(
        zxbcdt, xconv, delta, A_log, carry_a, carry_b);
    scan_pass2<<<G_HEADS, 256, 0, stream>>>(carry_a, carry_b, initst);
    scan_pass3<<<dim3(G_HEADS, NCHUNK), 256, 0, stream>>>(
        zxbcdt, xconv, delta, A_log, Dvec, initst, ygated);

    // 8) out = ygated @ W_out  (2048 x 2048 x 1024)
    swizzleA_bf16<<<(int)(SZ_ASW2 + 255) / 256, 256, 0, stream>>>(
        ygated, Asw2, L_SEQ/16, D_INNER/32, D_INNER);
    swizzleB_bf16<<<(int)(SZ_BSW2 + 255) / 256, 256, 0, stream>>>(
        W_out, Bsw2, D_MODEL/16, D_INNER/32, D_MODEL);
    gemm_bf16_wmma<<<dim3(L_SEQ/256, D_MODEL/64), 128, 0, stream>>>(
        Asw2, Bsw2, out, D_INNER/32, D_MODEL);
}
